// TalkingHeadAttention_32615981645966
// MI455X (gfx1250) — matrix operand order repaired and checked by reading, not compiled, not run on hardware
//
#include <hip/hip_runtime.h>
#include <hip/hip_bf16.h>

// B=2, N=2048, C=256, H=8, HD=32
#define SCALEQ 0.17677669529663687f  // 32^-0.5
#define NEGBIG -1e9f

typedef __bf16 bf16;
typedef __attribute__((ext_vector_type(8)))  bf16  v8bf;
typedef __attribute__((ext_vector_type(16))) bf16  v16bf;
typedef __attribute__((ext_vector_type(8)))  float v8f;

// A fragment (16x32, M x K): lane r = lane%16 is row M; element chunks
// K = [kb, kb+8) and [kb+16, kb+24) with kb = (lane<16 ? 0 : 8).
__device__ __forceinline__ v16bf load_a_frag(const bf16* base, int stride, int lane) {
  const int r  = lane & 15;
  const int kb = (lane & 16) ? 8 : 0;
  const bf16* p = base + r * stride + kb;
  v8bf lo = *(const v8bf*)(p);
  v8bf hi = *(const v8bf*)(p + 16);
  v16bf f;
#pragma unroll
  for (int i = 0; i < 8; ++i) { f[i] = lo[i]; f[i + 8] = hi[i]; }
  return f;
}

// B fragment (32x16, K x N): lane holds column N = lane%16; element chunks
// K = [kb, kb+8) and [kb+16, kb+24) with kb = (lane<16 ? 0 : 8). Source must be stored
// "column-major" (i.e. transposed so each N-column is a contiguous row).
__device__ __forceinline__ v16bf load_b_frag(const bf16* base, int stride, int lane) {
  const int r  = lane & 15;
  const int kb = (lane & 16) ? 8 : 0;
  const bf16* p = base + r * stride + kb;
  v8bf lo = *(const v8bf*)(p);
  v8bf hi = *(const v8bf*)(p + 16);
  v16bf f;
#pragma unroll
  for (int i = 0; i < 8; ++i) { f[i] = lo[i]; f[i + 8] = hi[i]; }
  return f;
}

__device__ __forceinline__ v8f wmma_bf16(v16bf a, v16bf b, v8f c) {
  return __builtin_amdgcn_wmma_f32_16x16x32_bf16(false, a, false, b, (short)0, c, false, false);
}

// ---------------- prep: f32 -> bf16 conversions ----------------
__global__ void k_conv_bf16(const float* __restrict__ in, bf16* __restrict__ out, int n) {
  int i = blockIdx.x * blockDim.x + threadIdx.x;
  if (i < n) out[i] = (bf16)in[i];
}

// out[b][c][r] = in[b][r][c]  (batched transpose + convert)
__global__ void k_transpose_bf16(const float* __restrict__ in, bf16* __restrict__ out,
                                 int R, int C, int total) {
  int i = blockIdx.x * blockDim.x + threadIdx.x;
  if (i >= total) return;
  int b = i / (R * C);
  int rem = i - b * R * C;
  int c = rem / R;
  int r = rem - c * R;
  out[i] = (bf16)in[(b * R + r) * C + c];
}

// ---------------- QKV projection: [4096,256] x [256,768] ----------------
// Scatters q (pre-scaled) -> Qraw[B,H,N,HD], k -> Kb[B,H,N,HD], v -> Vt[B,H,HD,N]
__global__ __launch_bounds__(256) void k_qkv(const bf16* __restrict__ xb,
                                             const bf16* __restrict__ wT,
                                             const float* __restrict__ bias,
                                             bf16* __restrict__ Qraw,
                                             bf16* __restrict__ Kb,
                                             bf16* __restrict__ Vt) {
  const int lane = threadIdx.x & 31;
  const int gid  = blockIdx.x * 8 + (threadIdx.x >> 5);  // 12288 tiles of 16x16
  const int mt = gid / 48, nt = gid - mt * 48;
  v8f acc = {};
#pragma unroll
  for (int kk = 0; kk < 256; kk += 32) {
    v16bf a = load_a_frag(xb + mt * 16 * 256 + kk, 256, lane);
    v16bf b = load_b_frag(wT + nt * 16 * 256 + kk, 256, lane);
    acc = wmma_bf16(a, b, acc);
  }
  const int col = nt * 16 + (lane & 15);
  const float bv = bias[col];
  const int which = col >> 8, h = (col >> 5) & 7, hd = col & 31;
  const int hi8 = (lane & 16) ? 8 : 0;
#pragma unroll
  for (int r = 0; r < 8; ++r) {
    const int row = mt * 16 + r + hi8;
    const int b0 = row >> 11, n = row & 2047;
    const float v = acc[r] + bv;
    if (which == 0)      Qraw[((b0 * 8 + h) * 2048 + n) * 32 + hd] = (bf16)(v * SCALEQ);
    else if (which == 1) Kb  [((b0 * 8 + h) * 2048 + n) * 32 + hd] = (bf16)v;
    else                 Vt  [((b0 * 8 + h) * 32 + hd) * 2048 + n] = (bf16)v;
  }
}

// ---------------- fold per-head interaction matrix: Qf = Qraw @ IM[h] ----------------
__global__ __launch_bounds__(256) void k_qim(const bf16* __restrict__ Qraw,
                                             const bf16* __restrict__ IMt,
                                             bf16* __restrict__ Qf) {
  const int lane = threadIdx.x & 31;
  const int gid = blockIdx.x * 8 + (threadIdx.x >> 5);  // 4096 tiles
  const int ntile = gid & 1;
  const int mtile = (gid >> 1) & 127;
  const int h  = (gid >> 8) & 7;
  const int b0 = gid >> 11;
  v16bf a  = load_a_frag(Qraw + ((b0 * 8 + h) * 2048 + mtile * 16) * 32, 32, lane);
  v16bf bB = load_b_frag(IMt + (h * 32 + ntile * 16) * 32, 32, lane);
  v8f acc = {};
  acc = wmma_bf16(a, bB, acc);
  const int hi8 = (lane & 16) ? 8 : 0;
  const int colb = ntile * 16 + (lane & 15);
#pragma unroll
  for (int r = 0; r < 8; ++r) {
    int rowm = mtile * 16 + r + hi8;
    Qf[((b0 * 8 + h) * 2048 + rowm) * 32 + colb] = (bf16)acc[r];
  }
}

// ---------------- fused talking-heads flash attention ----------------
// One WG = 32 queries x all 8 heads (wave w handles head w in the WMMA
// stages; thread t handles row (g = t/32, q = t%32) in the mixing stages).
// Two passes over the 64 key tiles: pass A collects softmax m/l stats,
// pass B recomputes scores, normalizes, applies post-softmax Ww mix and
// accumulates P'V with WMMA.
__global__ __launch_bounds__(256) void k_attn(const bf16* __restrict__ Qf,
                                              const bf16* __restrict__ Kb,
                                              const bf16* __restrict__ Vt,
                                              const unsigned char* __restrict__ mask,
                                              const float* __restrict__ Wl,
                                              const float* __restrict__ bl,
                                              const float* __restrict__ Ww,
                                              const float* __restrict__ bw,
                                              bf16* __restrict__ ctx) {
  __shared__ float Sraw[8][32][33];   // raw per-head scores (padded vs bank conflicts)
  __shared__ float Pn[8][32][33];     // normalized probabilities
  __shared__ bf16  Pmix[8][32][40];   // post-mixed probs, bf16, 80B rows (16B-aligned, conflict-free)

  const int tid  = threadIdx.x;
  const int lane = tid & 31;
  const int wv   = tid >> 5;
  const int q    = lane;
  const int b0   = blockIdx.x >> 6;
  const int qbase = (blockIdx.x & 63) * 32;

  v16bf aQ0 = load_a_frag(Qf + ((b0 * 8 + wv) * 2048 + qbase) * 32, 32, lane);
  v16bf aQ1 = load_a_frag(Qf + ((b0 * 8 + wv) * 2048 + qbase + 16) * 32, 32, lane);

  float wl[8], ww[8];
#pragma unroll
  for (int hh = 0; hh < 8; ++hh) {
    wl[hh] = Wl[hh * 8 + wv];   // Wl[h][g], this thread's g = wv
    ww[hh] = Ww[hh * 8 + wv];   // Ww[g][g'], this thread's g' = wv
  }
  const float blv = bl[wv];
  const float bwv = bw[wv];
  const unsigned char* mrow = mask + b0 * 2048;

  const int hi8  = (lane & 16) ? 8 : 0;
  const int ncol = lane & 15;
  const bf16* Kbase = Kb + (b0 * 8 + wv) * 2048 * 32;

  float m_run = -INFINITY, l_run = 0.0f;

  // ----- pass A: softmax statistics -----
  for (int kt = 0; kt < 64; ++kt) {
    v8f s00 = {}, s01 = {}, s10 = {}, s11 = {};
    v16bf bK0 = load_b_frag(Kbase + (kt * 32) * 32, 32, lane);
    v16bf bK1 = load_b_frag(Kbase + (kt * 32 + 16) * 32, 32, lane);
    s00 = wmma_bf16(aQ0, bK0, s00);
    s01 = wmma_bf16(aQ0, bK1, s01);
    s10 = wmma_bf16(aQ1, bK0, s10);
    s11 = wmma_bf16(aQ1, bK1, s11);
#pragma unroll
    for (int r = 0; r < 8; ++r) {
      Sraw[wv][r + hi8][ncol]            = s00[r];
      Sraw[wv][r + hi8][16 + ncol]       = s01[r];
      Sraw[wv][16 + r + hi8][ncol]       = s10[r];
      Sraw[wv][16 + r + hi8][16 + ncol]  = s11[r];
    }
    __syncthreads();
#pragma unroll 4
    for (int k = 0; k < 32; ++k) {
      float s = blv;
#pragma unroll
      for (int hh = 0; hh < 8; ++hh) s += Sraw[hh][q][k] * wl[hh];
      if (!mrow[kt * 32 + k]) s += NEGBIG;
      float mnew = fmaxf(m_run, s);
      l_run = l_run * __expf(m_run - mnew) + __expf(s - mnew);
      m_run = mnew;
    }
    __syncthreads();
  }
  const float invl = 1.0f / l_run;

  // ----- pass B: normalized probs, post-mix, PV -----
  v8f o00 = {}, o01 = {}, o10 = {}, o11 = {};
  for (int kt = 0; kt < 64; ++kt) {
    v8f s00 = {}, s01 = {}, s10 = {}, s11 = {};
    v16bf bK0 = load_b_frag(Kbase + (kt * 32) * 32, 32, lane);
    v16bf bK1 = load_b_frag(Kbase + (kt * 32 + 16) * 32, 32, lane);
    s00 = wmma_bf16(aQ0, bK0, s00);
    s01 = wmma_bf16(aQ0, bK1, s01);
    s10 = wmma_bf16(aQ1, bK0, s10);
    s11 = wmma_bf16(aQ1, bK1, s11);
#pragma unroll
    for (int r = 0; r < 8; ++r) {
      Sraw[wv][r + hi8][ncol]            = s00[r];
      Sraw[wv][r + hi8][16 + ncol]       = s01[r];
      Sraw[wv][16 + r + hi8][ncol]       = s10[r];
      Sraw[wv][16 + r + hi8][16 + ncol]  = s11[r];
    }
    __syncthreads();
#pragma unroll 4
    for (int k = 0; k < 32; ++k) {
      float s = blv;
#pragma unroll
      for (int hh = 0; hh < 8; ++hh) s += Sraw[hh][q][k] * wl[hh];
      if (!mrow[kt * 32 + k]) s += NEGBIG;
      Pn[wv][q][k] = __expf(s - m_run) * invl;
    }
    __syncthreads();
#pragma unroll 4
    for (int k = 0; k < 32; ++k) {
      float pm = bwv;
#pragma unroll
      for (int g = 0; g < 8; ++g) pm += Pn[g][q][k] * ww[g];
      Pmix[wv][q][k] = (bf16)pm;
    }
    __syncthreads();
    {
      v16bf aP0 = load_a_frag(&Pmix[wv][0][0], 40, lane);
      v16bf aP1 = load_a_frag(&Pmix[wv][16][0], 40, lane);
      const bf16* Vbase = Vt + (b0 * 8 + wv) * 32 * 2048 + kt * 32;
      v16bf bV0 = load_b_frag(Vbase, 2048, lane);
      v16bf bV1 = load_b_frag(Vbase + 16 * 2048, 2048, lane);
      o00 = wmma_bf16(aP0, bV0, o00);
      o01 = wmma_bf16(aP0, bV1, o01);
      o10 = wmma_bf16(aP1, bV0, o10);
      o11 = wmma_bf16(aP1, bV1, o11);
    }
    __syncthreads();
  }
  // merge heads into ctx[b][n][wv*32 + c]
#pragma unroll
  for (int r = 0; r < 8; ++r) {
    int n0 = qbase + r + hi8;
    int n1 = qbase + 16 + r + hi8;
    ctx[(b0 * 2048 + n0) * 256 + wv * 32 + ncol]      = (bf16)o00[r];
    ctx[(b0 * 2048 + n0) * 256 + wv * 32 + 16 + ncol] = (bf16)o01[r];
    ctx[(b0 * 2048 + n1) * 256 + wv * 32 + ncol]      = (bf16)o10[r];
    ctx[(b0 * 2048 + n1) * 256 + wv * 32 + 16 + ncol] = (bf16)o11[r];
  }
}

// ---------------- output projection: ctx [4096,256] x Wp [256,256] + bp ----------------
__global__ __launch_bounds__(256) void k_proj(const bf16* __restrict__ ctx,
                                              const bf16* __restrict__ wT,
                                              const float* __restrict__ bias,
                                              float* __restrict__ out) {
  const int lane = threadIdx.x & 31;
  const int gid = blockIdx.x * 8 + (threadIdx.x >> 5);  // 4096 tiles: 256 x 16
  const int mt = gid >> 4, nt = gid & 15;
  v8f acc = {};
#pragma unroll
  for (int kk = 0; kk < 256; kk += 32) {
    v16bf a = load_a_frag(ctx + mt * 16 * 256 + kk, 256, lane);
    v16bf b = load_b_frag(wT + nt * 16 * 256 + kk, 256, lane);
    acc = wmma_bf16(a, b, acc);
  }
  const int col = nt * 16 + (lane & 15);
  const float bv = bias[col];
  const int hi8 = (lane & 16) ? 8 : 0;
#pragma unroll
  for (int r = 0; r < 8; ++r) {
    int row = mt * 16 + r + hi8;
    out[row * 256 + col] = acc[r] + bv;
  }
}

extern "C" void kernel_launch(void* const* d_in, const int* in_sizes, int n_in,
                              void* d_out, int out_size, void* d_ws, size_t ws_size,
                              hipStream_t stream) {
  (void)in_sizes; (void)n_in; (void)out_size; (void)ws_size;
  const float* x            = (const float*)d_in[0];
  const unsigned char* mask = (const unsigned char*)d_in[1];
  const float* Wqkv         = (const float*)d_in[2];
  const float* bqkv         = (const float*)d_in[3];
  const float* Wl           = (const float*)d_in[4];
  const float* bl           = (const float*)d_in[5];
  const float* Ww           = (const float*)d_in[6];
  const float* bw           = (const float*)d_in[7];
  const float* Wp           = (const float*)d_in[8];
  const float* bp           = (const float*)d_in[9];
  const float* IM           = (const float*)d_in[10];
  float* out = (float*)d_out;

  char* w = (char*)d_ws;
  bf16* xb    = (bf16*)(w);              // [4096][256]        2 MB
  bf16* WqkvT = (bf16*)(w + 2097152);    // [768][256]         384 KB
  bf16* WpT   = (bf16*)(w + 2490368);    // [256][256]         128 KB
  bf16* IMt   = (bf16*)(w + 2621440);    // [8][32][32]        16 KB
  bf16* Qraw  = (bf16*)(w + 2637824);    // [2][8][2048][32]   2 MB
  bf16* Qf    = (bf16*)(w + 4734976);    // [2][8][2048][32]   2 MB
  bf16* Kb    = (bf16*)(w + 6832128);    // [2][8][2048][32]   2 MB
  bf16* Vt    = (bf16*)(w + 8929280);    // [2][8][32][2048]   2 MB
  bf16* ctx   = (bf16*)(w + 11026432);   // [4096][256]        2 MB

  k_conv_bf16<<<4096, 256, 0, stream>>>(x, xb, 4096 * 256);
  k_transpose_bf16<<<768, 256, 0, stream>>>(Wqkv, WqkvT, 256, 768, 256 * 768);
  k_transpose_bf16<<<256, 256, 0, stream>>>(Wp, WpT, 256, 256, 256 * 256);
  k_transpose_bf16<<<32, 256, 0, stream>>>(IM, IMt, 32, 32, 8 * 32 * 32);
  k_qkv<<<1536, 256, 0, stream>>>(xb, WqkvT, bqkv, Qraw, Kb, Vt);
  k_qim<<<512, 256, 0, stream>>>(Qraw, IMt, Qf);
  k_attn<<<128, 256, 0, stream>>>(Qf, Kb, Vt, mask, Wl, bl, Ww, bw, ctx);
  k_proj<<<512, 256, 0, stream>>>(ctx, WpT, bp, out);
}
